// GPT_63685775065100
// MI455X (gfx1250) — matrix operand order repaired and checked by reading, not compiled, not run on hardware
//
#include <hip/hip_runtime.h>

// ---------------------------------------------------------------------------
// GPT forward for MI455X (gfx1250, wave32). All GEMMs + attention use
// v_wmma_f32_16x16x32_f16 (f16 inputs, f32 accumulate). Global->LDS staging
// uses CDNA5 async-to-LDS DMA (global_load_async_to_lds_b128, ASYNCcnt) with
// double-buffered LDS so WMMA overlaps the copies. Working set fits in the
// 192MB L2, so the target is the WMMA ceiling, not HBM bandwidth.
// ---------------------------------------------------------------------------

typedef __attribute__((ext_vector_type(16))) _Float16 v16h;
typedef __attribute__((ext_vector_type(8)))  float    v8f;

union FragU { v16h v; unsigned int u[8]; _Float16 h[16]; };

__device__ __forceinline__ v8f wmma_f16(v16h a, v16h b, v8f c) {
  // 8 args: (neg_a, A, neg_b, B, c_mod, C, reuse_a, reuse_b)
  return __builtin_amdgcn_wmma_f32_16x16x32_f16(false, a, false, b, (short)0, c,
                                                false, false);
}

// CDNA5 async DMA: 16 bytes per lane, global -> LDS, tracked by ASYNCcnt.
__device__ __forceinline__ void async_b128(_Float16* lds_dst,
                                           const _Float16* gsrc) {
  unsigned lds_off = (unsigned)(unsigned long long)(
      __attribute__((address_space(3))) _Float16*)lds_dst;
  asm volatile("global_load_async_to_lds_b128 %0, %1, off" ::"v"(lds_off),
               "v"((unsigned long long)gsrc)
               : "memory");
}
__device__ __forceinline__ void wait_async0() {
  asm volatile("s_wait_asynccnt 0x0" ::: "memory");
}

__device__ __forceinline__ float rmax16(float v) {
#pragma unroll
  for (int m = 1; m < 16; m <<= 1) v = fmaxf(v, __shfl_xor(v, m, 32));
  return v;
}
__device__ __forceinline__ float rsum16(float v) {
#pragma unroll
  for (int m = 1; m < 16; m <<= 1) v += __shfl_xor(v, m, 32);
  return v;
}

// ---------------------------------------------------------------------------
// f32 -> f16 weight conversion
// ---------------------------------------------------------------------------
__global__ __launch_bounds__(256) void f32_to_f16(const float* __restrict__ s,
                                                  _Float16* __restrict__ d,
                                                  int n) {
  int i = blockIdx.x * 256 + threadIdx.x;
  if (i < n) d[i] = (_Float16)s[i];
}

// ---------------------------------------------------------------------------
// Embedding gather: x = tok_emb[idx] + pos_emb  (writes f32 and f16)
// fixed sizes: M = B*T = 8192, C = 256, T = 2048
// ---------------------------------------------------------------------------
__global__ __launch_bounds__(256) void embed_kernel(
    const int* __restrict__ idx, const float* __restrict__ tok,
    const float* __restrict__ pos, float* __restrict__ xf,
    _Float16* __restrict__ xh) {
  const int i = blockIdx.x * 256 + threadIdx.x;  // < 8192*256
  const int m = i >> 8;
  const int c = i & 255;
  const int t = m & 2047;  // T = 2048
  const float v = tok[(size_t)idx[m] * 256 + c] + pos[(size_t)t * 256 + c];
  xf[i] = v;
  xh[i] = (_Float16)v;
}

// ---------------------------------------------------------------------------
// WMMA GEMM: out[M,N] = A[M,K](f16) @ W[K,N](f16) [+ bias] [+ resid]
// Workgroup tile 128x64, 8 waves as 4x2, each wave 2x2 of 16x16 tiles.
// Double-buffered LDS; A tile staged with async-to-LDS DMA.
// Epilogue behavior is compile-time templated (no per-element branches).
// M%128==0, N%64==0, K%32==0 (always true for this model).
// ---------------------------------------------------------------------------
template <bool BIAS, bool RES, bool OUTF, bool OUTH>
__global__ __launch_bounds__(256) void gemm_ws(
    const _Float16* __restrict__ A, const _Float16* __restrict__ W,
    const float* __restrict__ bias, const float* __restrict__ resid,
    float* __restrict__ outF, _Float16* __restrict__ outH, int M, int N,
    int K) {
  __shared__ alignas(32) _Float16 Al[2][128 * 32];  // [m][k]
  __shared__ alignas(32) _Float16 Wt[2][64 * 32];   // [n][k] (transposed)

  const int tid = threadIdx.x;
  const int lane = tid & 31;
  const int wave = tid >> 5;
  const int half = lane >> 4;
  const int l15 = lane & 15;
  const int m0 = blockIdx.y * 128;
  const int n0 = blockIdx.x * 64;
  const int mw = (wave & 3) * 32;
  const int nw = (wave >> 2) * 32;

  // staging roles
  const int ra = tid >> 1;        // A: row within tile
  const int ca = (tid & 1) * 16;  // A: col (halves)
  const int kw = tid >> 3;        // W: k within tile
  const int nc = (tid & 7) * 8;   // W: n start (8 halves per thread)

  const v8f vz = {0.f, 0.f, 0.f, 0.f, 0.f, 0.f, 0.f, 0.f};
  v8f acc[2][2];
#pragma unroll
  for (int i = 0; i < 2; ++i)
#pragma unroll
    for (int j = 0; j < 2; ++j) acc[i][j] = vz;

  const _Float16* Arow = A + (size_t)(m0 + ra) * K;

  // ---- prologue: stage k-tile 0 into buffer 0 ----
  async_b128(&Al[0][ra * 32 + ca], Arow + ca);
  async_b128(&Al[0][ra * 32 + ca + 8], Arow + ca + 8);
  {
    const _Float16* g = W + (size_t)kw * N + n0 + nc;
    _Float16 t[8];
#pragma unroll
    for (int j = 0; j < 8; ++j) t[j] = g[j];
#pragma unroll
    for (int j = 0; j < 8; ++j) Wt[0][(nc + j) * 32 + kw] = t[j];
  }
  wait_async0();
  __syncthreads();

  int buf = 0;
  for (int k0 = 0; k0 < K; k0 += 32) {
    const int nb = buf ^ 1;
    const bool more = (k0 + 32) < K;
    _Float16 tw[8];
    if (more) {
      // kick off next tile: A via async DMA, W global loads into regs
      async_b128(&Al[nb][ra * 32 + ca], Arow + k0 + 32 + ca);
      async_b128(&Al[nb][ra * 32 + ca + 8], Arow + k0 + 32 + ca + 8);
      const _Float16* g = W + (size_t)(k0 + 32 + kw) * N + n0 + nc;
#pragma unroll
      for (int j = 0; j < 8; ++j) tw[j] = g[j];
    }

    // ---- compute on current buffer ----
    FragU a[2];
#pragma unroll
    for (int ti = 0; ti < 2; ++ti) {
      const _Float16* src = &Al[buf][(mw + ti * 16 + l15) * 32];
#pragma unroll
      for (int v = 0; v < 8; ++v) {
        const int kk = half * 16 + 2 * v;
        a[ti].u[v] = *(const unsigned int*)(src + kk);
      }
    }
#pragma unroll
    for (int tj = 0; tj < 2; ++tj) {
      v16h b = *(const v16h*)(&Wt[buf][(nw + tj * 16 + l15) * 32 + half * 16]);
#pragma unroll
      for (int ti = 0; ti < 2; ++ti)
        acc[ti][tj] = wmma_f16(a[ti].v, b, acc[ti][tj]);
    }

    if (more) {
      // transpose-scatter W regs into next buffer, then drain DMA
#pragma unroll
      for (int j = 0; j < 8; ++j) Wt[nb][(nc + j) * 32 + kw] = tw[j];
      wait_async0();
    }
    __syncthreads();
    buf = nb;
  }

  // epilogue: D layout row = half*8 + r, col = l15 (branch-free, templated)
#pragma unroll
  for (int ti = 0; ti < 2; ++ti)
#pragma unroll
    for (int tj = 0; tj < 2; ++tj) {
      const int gn = n0 + nw + tj * 16 + l15;
      float bb = 0.f;
      if (BIAS) bb = bias[gn];
#pragma unroll
      for (int r = 0; r < 8; ++r) {
        const int gm = m0 + mw + ti * 16 + half * 8 + r;
        const size_t o = (size_t)gm * N + gn;
        float v = acc[ti][tj][r] + bb;
        if (RES) v += resid[o];
        if (OUTF) outF[o] = v;
        if (OUTH) outH[o] = (_Float16)v;
      }
    }
}

// ---------------------------------------------------------------------------
// Flash attention with WMMA. qkv: [B*T, 768] f16 (q|k|v each 256 cols,
// head h at col h*64). One block = 4 waves = 64 query rows of one (b,h).
// Key chunks of 32; causal. Q/K staged via async-to-LDS DMA. y: [B*T,256] f16
// ---------------------------------------------------------------------------
__global__ __launch_bounds__(128) void attn_flash(
    const _Float16* __restrict__ qkv, _Float16* __restrict__ y) {
  constexpr int T = 2048, D = 64, C3 = 768, C = 256;
  __shared__ alignas(32) _Float16 Ql[64 * 64];      // [q][d]
  __shared__ alignas(32) _Float16 Kl[32 * 64];      // [key][d]
  __shared__ alignas(32) _Float16 Vt[64 * 32];      // [d][key]
  __shared__ alignas(32) _Float16 Pl[4 * 16 * 32];  // per-wave P tile

  const int tid = threadIdx.x;
  const int lane = tid & 31, wave = tid >> 5;
  const int half = lane >> 4, l15 = lane & 15;

  const int qt = blockIdx.x & 31;       // T/64 = 32 query tiles
  const int h = (blockIdx.x >> 5) & 3;  // H = 4
  const int b = blockIdx.x >> 7;
  const int q0 = qt * 64;
  const size_t rowbase = (size_t)(b * T) * C3;

  {  // stage Q tile 64x64 via async DMA (64B per thread)
    const int r = tid >> 1;
    const int c = (tid & 1) * 32;
    const _Float16* g = qkv + rowbase + (size_t)(q0 + r) * C3 + h * D + c;
    _Float16* l = Ql + r * 64 + c;
#pragma unroll
    for (int j = 0; j < 4; ++j) async_b128(l + j * 8, g + j * 8);
  }
  wait_async0();
  __syncthreads();

  // Q A-fragments for both 32-wide contraction chunks of D=64
  FragU qa[2];
#pragma unroll
  for (int c = 0; c < 2; ++c) {
    const _Float16* src = Ql + (wave * 16 + l15) * 64 + c * 32;
#pragma unroll
    for (int v = 0; v < 8; ++v) {
      const int kk = half * 16 + 2 * v;
      qa[c].u[v] = *(const unsigned int*)(src + kk);
    }
  }

  const v8f vz = {0.f, 0.f, 0.f, 0.f, 0.f, 0.f, 0.f, 0.f};
  v8f yacc[4];
#pragma unroll
  for (int d = 0; d < 4; ++d) yacc[d] = vz;
  float mrun[8], lrun[8];
#pragma unroll
  for (int r = 0; r < 8; ++r) {
    mrun[r] = -1e30f;
    lrun[r] = 0.f;
  }

  const int nchunk = q0 / 32 + 2;  // keys [0, q0+64)
  const int qhi = q0 + wave * 16 + 15;

  for (int j = 0; j < nchunk; ++j) {
    {  // stage K [32x64] async; V transposed [64x32] via regs
      const int key = tid >> 2;
      const int c = (tid & 3) * 16;
      const _Float16* gk =
          qkv + rowbase + (size_t)(j * 32 + key) * C3 + 256 + h * D + c;
      async_b128(Kl + key * 64 + c, gk);
      async_b128(Kl + key * 64 + c + 8, gk + 8);
      const _Float16* gv =
          qkv + rowbase + (size_t)(j * 32 + key) * C3 + 512 + h * D + c;
      v16h tv = *(const v16h*)gv;
#pragma unroll
      for (int i = 0; i < 16; ++i) Vt[(c + i) * 32 + key] = tv[i];
    }
    wait_async0();
    __syncthreads();

    const bool active = (j * 32) <= qhi;
    if (active) {
      // S = Q @ K^T : two 16-key column tiles, contraction D split in 2
      v8f s[2];
      s[0] = vz;
      s[1] = vz;
#pragma unroll
      for (int nt = 0; nt < 2; ++nt) {
#pragma unroll
        for (int c = 0; c < 2; ++c) {
          v16h bK =
              *(const v16h*)(Kl + (nt * 16 + l15) * 64 + c * 32 + half * 16);
          s[nt] = wmma_f16(qa[c].v, bK, s[nt]);
        }
      }
      // online softmax (row lives in 16 lanes of this half-wave)
      const int qrow0 = q0 + wave * 16;
      const int keyb = j * 32 + l15;
#pragma unroll
      for (int r = 0; r < 8; ++r) {
        const int rq = qrow0 + half * 8 + r;
        float s0 = s[0][r] * 0.125f;  // 1/sqrt(64)
        float s1 = s[1][r] * 0.125f;
        if (keyb > rq) s0 = -1e30f;
        if (keyb + 16 > rq) s1 = -1e30f;
        const float mx = rmax16(fmaxf(s0, s1));
        const float mn = fmaxf(mrun[r], mx);
        const float p0 = __expf(s0 - mn);
        const float p1 = __expf(s1 - mn);
        const float rs = rsum16(p0 + p1);
        const float alpha = __expf(mrun[r] - mn);
        mrun[r] = mn;
        lrun[r] = lrun[r] * alpha + rs;
#pragma unroll
        for (int d = 0; d < 4; ++d) yacc[d][r] = yacc[d][r] * alpha;
        _Float16* pr = Pl + wave * 512 + (half * 8 + r) * 32;
        pr[l15] = (_Float16)p0;
        pr[16 + l15] = (_Float16)p1;
      }
    }
    __syncthreads();  // make P visible within the wave's LDS region
    if (active) {
      FragU pa;
      const _Float16* src = Pl + wave * 512 + l15 * 32;
#pragma unroll
      for (int v = 0; v < 8; ++v) {
        const int kk = half * 16 + 2 * v;
        pa.u[v] = *(const unsigned int*)(src + kk);
      }
#pragma unroll
      for (int d = 0; d < 4; ++d) {
        v16h bV = *(const v16h*)(Vt + (d * 16 + l15) * 32 + half * 16);
        yacc[d] = wmma_f16(pa.v, bV, yacc[d]);
      }
    }
    __syncthreads();  // protect K/Vt before next chunk's staging
  }

  // write y (f16), normalized by row sums
#pragma unroll
  for (int d = 0; d < 4; ++d) {
#pragma unroll
    for (int r = 0; r < 8; ++r) {
      const int row = q0 + wave * 16 + half * 8 + r;
      const float v = yacc[d][r] / lrun[r];
      y[(size_t)(b * T + row) * C + h * D + d * 16 + l15] = (_Float16)v;
    }
  }
}

// ---------------------------------------------------------------------------
// Host orchestration
// ---------------------------------------------------------------------------
extern "C" void kernel_launch(void* const* d_in, const int* in_sizes, int n_in,
                              void* d_out, int out_size, void* d_ws,
                              size_t ws_size, hipStream_t stream) {
  (void)in_sizes;
  (void)n_in;
  (void)out_size;
  (void)ws_size;
  const int* idx = (const int*)d_in[0];
  const float* tok = (const float*)d_in[1];
  const float* pos = (const float*)d_in[2];
  const float* Wqkv = (const float*)d_in[3];
  const float* bqkv = (const float*)d_in[4];
  const float* Wo = (const float*)d_in[5];
  const float* bo = (const float*)d_in[6];
  const float* Wfc = (const float*)d_in[7];
  const float* bfc = (const float*)d_in[8];
  const float* Wfp = (const float*)d_in[9];
  const float* bfp = (const float*)d_in[10];
  const float* Whead = (const float*)d_in[11];
  float* out = (float*)d_out;

  const int M = 8192;  // B*T
  char* base = (char*)d_ws;
  size_t off = 0;
  auto take = [&](size_t elems) -> _Float16* {
    _Float16* p = (_Float16*)(base + off);
    off += ((elems * sizeof(_Float16)) + 255) & ~(size_t)255;
    return p;
  };
  _Float16* hWqkv = take((size_t)4 * 256 * 768);
  _Float16* hWo = take((size_t)4 * 256 * 256);
  _Float16* hWfc = take((size_t)4 * 256 * 1024);
  _Float16* hWfp = take((size_t)4 * 1024 * 256);
  _Float16* hWhead = take((size_t)256 * 2048);
  _Float16* xh = take((size_t)M * 256);
  _Float16* qkvh = take((size_t)M * 768);
  _Float16* yh = take((size_t)M * 256);
  _Float16* hh = take((size_t)M * 1024);
  float* xf = (float*)(base + off);
  off += (size_t)M * 256 * sizeof(float);

  auto conv = [&](const float* s, _Float16* d, int n) {
    f32_to_f16<<<(n + 255) / 256, 256, 0, stream>>>(s, d, n);
  };
  conv(Wqkv, hWqkv, 4 * 256 * 768);
  conv(Wo, hWo, 4 * 256 * 256);
  conv(Wfc, hWfc, 4 * 256 * 1024);
  conv(Wfp, hWfp, 4 * 1024 * 256);
  conv(Whead, hWhead, 256 * 2048);

  embed_kernel<<<(M * 256) / 256, 256, 0, stream>>>(idx, tok, pos, xf, xh);

  for (int l = 0; l < 4; ++l) {
    // qkv = x @ Wqkv + b  (f16 only; consumed by attention)
    gemm_ws<true, false, false, true>
        <<<dim3(768 / 64, M / 128), 256, 0, stream>>>(
            xh, hWqkv + (size_t)l * 256 * 768, bqkv + l * 768, nullptr,
            nullptr, qkvh, M, 768, 256);
    attn_flash<<<512, 128, 0, stream>>>(qkvh, yh);
    // x = x + y @ Wo + b  (f32 residual update + f16 copy)
    gemm_ws<true, true, true, true>
        <<<dim3(256 / 64, M / 128), 256, 0, stream>>>(
            yh, hWo + (size_t)l * 256 * 256, bo + l * 256, xf, xf, xh, M, 256,
            256);
    // h = x @ Wfc + b (no activation in source model)
    gemm_ws<true, false, false, true>
        <<<dim3(1024 / 64, M / 128), 256, 0, stream>>>(
            xh, hWfc + (size_t)l * 256 * 1024, bfc + l * 1024, nullptr,
            nullptr, hh, M, 1024, 256);
    // x = x + h @ Wfp + b
    gemm_ws<true, true, true, true>
        <<<dim3(256 / 64, M / 128), 256, 0, stream>>>(
            hh, hWfp + (size_t)l * 1024 * 256, bfp + l * 256, xf, xf, xh, M,
            256, 1024);
  }
  // logits = x @ Whead (f32 out, no bias)
  gemm_ws<false, false, true, false>
      <<<dim3(2048 / 64, M / 128), 256, 0, stream>>>(
          xh, hWhead, nullptr, nullptr, out, nullptr, M, 2048, 256);
}
